// MultiHeadQueryInstanceDecoder_89756226551892
// MI455X (gfx1250) — matrix operand order repaired and checked by reading, not compiled, not run on hardware
//
#include <hip/hip_runtime.h>
#include <hip/hip_bf16.h>
#include <math.h>

// ---------------- constants ----------------
#define NN   262144   // points
#define CC   72       // in channels
#define CP   96       // padded in channels (multiple of 32 for WMMA K)
#define DD   256      // hidden
#define GQ   384      // G*Q
#define NBLK 512      // n-blocks in logits kernel (512 cols each)
#define KSPL 64       // split-K factor for pooling
#define LDF  264      // padded f32 LDS row stride
#define LDH  264      // padded f16 LDS row stride

typedef __attribute__((ext_vector_type(16))) _Float16 v16h;
typedef __attribute__((ext_vector_type(8)))  float    v8f;

__device__ __forceinline__ v8f wmma_f16(v16h a, v16h b, v8f c) {
  return __builtin_amdgcn_wmma_f32_16x16x32_f16(false, a, false, b, (short)0, c, false, false);
}

__device__ __forceinline__ float gelu_erf(float x) {
  return 0.5f * x * (1.0f + erff(x * 0.70710678118654752440f));
}

// A fragment (16x32 f16): per ISA layout, lane = (m = lane&15, half = lane>>4);
// element e<8 -> k = kb + 8*half + e ; e>=8 -> k = kb + 16 + 8*half + (e-8)
__device__ __forceinline__ v16h ldA_f16(const _Float16* __restrict__ p, int ld, int row, int kb, int half) {
  v16h a;
  const _Float16* r0 = p + (size_t)row * ld + kb + 8 * half;
  const _Float16* r1 = r0 + 16;
#pragma unroll
  for (int e = 0; e < 8; e++) { a[e] = r0[e]; a[8 + e] = r1[e]; }
  return a;
}

// B fragment (32x16 f16): element (k,col) at p[col*ld + k]; lane = (n = lane&15,
// half = lane>>4); element e<8 -> k = kb + 8*half + e ; e>=8 -> k = kb + 16 + 8*half + (e-8)
__device__ __forceinline__ v16h ldB_T(const _Float16* __restrict__ p, int ld, int col, int kb, int half) {
  v16h b;
  const _Float16* r = p + (size_t)col * ld + kb + 8 * half;
#pragma unroll
  for (int e = 0; e < 8; e++) { b[e] = r[e]; b[8 + e] = r[16 + e]; }
  return b;
}

// Pooling A fragment: w(row,k) = exp(logit[row, nb+k] - sg) * norms[nb+k], as f16
__device__ __forceinline__ v16h ldA_w(const float* __restrict__ lg, int ldn, int row, int nb, int half,
                                      const float* __restrict__ norms, float sg) {
  v16h a;
  const float* base = lg + (size_t)row * ldn + nb;
#pragma unroll
  for (int e = 0; e < 8; e++) {
    int k0 = 8 * half + e;
    int k1 = k0 + 16;
    a[e]     = (_Float16)(__expf(base[k0] - sg) * norms[nb + k0]);
    a[8 + e] = (_Float16)(__expf(base[k1] - sg) * norms[nb + k1]);
  }
  return a;
}

// ---------------- weight prep: transpose + f16 convert ----------------
__global__ __launch_bounds__(256) void k_prep(const float* __restrict__ w1, const float* __restrict__ w2,
                                              const float* __restrict__ p1, const float* __restrict__ p2,
                                              _Float16* __restrict__ w1T, _Float16* __restrict__ w2T,
                                              _Float16* __restrict__ p1T, _Float16* __restrict__ p2T) {
  int i = blockIdx.x * 256 + threadIdx.x;
  if (i < 256 * CP) {
    int nn = i / CP, k = i % CP;
    w1T[i] = (k < CC) ? (_Float16)w1[k * DD + nn] : (_Float16)0.f;
  }
  if (i < 65536) {
    int nn = i >> 8, k = i & 255;
    w2T[i] = (_Float16)w2[k * DD + nn];
    p1T[i] = (_Float16)p1[k * DD + nn];
    p2T[i] = (_Float16)p2[k * DD + nn];
  }
}

// ---------------- point_feat pad+convert: [N,72] f32 -> [N,96] f16 (zeros beyond 72) ----------------
__global__ __launch_bounds__(256) void k_prep_pf(const float* __restrict__ pf, _Float16* __restrict__ pf16) {
  size_t i = (size_t)blockIdx.x * 256 + threadIdx.x;   // over N*96 outputs
  int k = (int)(i % CP);
  size_t row = i / CP;
  pf16[i] = (k < CC) ? (_Float16)pf[row * CC + k] : (_Float16)0.f;
}

// ---------------- deterministic in-block sum ----------------
__device__ __forceinline__ float blk_sum(float v, float* rs) {
  int t = threadIdx.x;
  rs[t] = v;
  __syncthreads();
  for (int s = 128; s > 0; s >>= 1) { if (t < s) rs[t] += rs[t + s]; __syncthreads(); }
  float r = rs[0];
  __syncthreads();
  return r;
}

// ---------------- query MLP: query_embed + normalized f16 qmask ----------------
__global__ __launch_bounds__(256) void k_query(const float* __restrict__ qe,
                                               const float* __restrict__ lng, const float* __restrict__ lnb,
                                               const float* __restrict__ w1, const float* __restrict__ b1,
                                               const float* __restrict__ w2, const float* __restrict__ b2,
                                               float* __restrict__ qemb, _Float16* __restrict__ qm16) {
  __shared__ float x[256], y[256], rs[256];
  int gq = blockIdx.x, g = gq >> 7, t = threadIdx.x;
  float xe = qe[(size_t)gq * DD + t];
  float m  = blk_sum(xe, rs) * (1.f / 256.f);
  float dv = xe - m;
  float v  = blk_sum(dv * dv, rs) * (1.f / 256.f);
  float xn = dv * rsqrtf(v + 1e-5f) * lng[g * DD + t] + lnb[g * DD + t];
  x[t] = xn;
  __syncthreads();
  float a = b1[g * DD + t];
  const float* wp = w1 + (size_t)g * DD * DD + t;
  for (int d = 0; d < DD; d++) a += x[d] * wp[(size_t)d * DD];
  y[t] = gelu_erf(a);
  __syncthreads();
  float z = b2[g * DD + t];
  const float* wp2 = w2 + (size_t)g * DD * DD + t;
  for (int d = 0; d < DD; d++) z += y[d] * wp2[(size_t)d * DD];
  z += xe;  // residual
  qemb[(size_t)gq * DD + t] = z;
  float nr = sqrtf(blk_sum(z * z, rs));
  nr = fmaxf(nr, 1e-12f);
  qm16[(size_t)gq * DD + t] = (_Float16)(z / nr);
}

// ---------------- LN stats over 16x256 tile in LDS ----------------
__device__ __forceinline__ void ln_stats(const float* sf, float* red1, float* red2,
                                         float* rowm, float* rowr) {
  int tid = threadIdx.x;
  int row = tid >> 4, seg = tid & 15;
  const float* p = &sf[row * LDF + seg * 16];
  float s = 0.f, q = 0.f;
#pragma unroll
  for (int i = 0; i < 16; i++) { float x = p[i]; s += x; q += x * x; }
  red1[tid] = s; red2[tid] = q;
  __syncthreads();
  if (tid < 16) {
    float ss = 0.f, qq = 0.f;
#pragma unroll
    for (int i = 0; i < 16; i++) { ss += red1[tid * 16 + i]; qq += red2[tid * 16 + i]; }
    float m = ss * (1.f / 256.f);
    float vv = qq * (1.f / 256.f) - m * m;
    rowm[tid] = m;
    rowr[tid] = rsqrtf(vv + 1e-5f);
  }
  __syncthreads();
}

// ---------------- fused point trunk: 4 chained WMMA GEMMs per 16-row tile ----------------
__global__ __launch_bounds__(256) void k_trunk(
    const _Float16* __restrict__ pf16,
    const _Float16* __restrict__ w1T, const float* __restrict__ b1,
    const float* __restrict__ g1, const float* __restrict__ be1,
    const _Float16* __restrict__ w2T, const float* __restrict__ b2,
    const float* __restrict__ phg, const float* __restrict__ phb,
    const _Float16* __restrict__ pw1T, const float* __restrict__ pb1v,
    const _Float16* __restrict__ pw2T, const float* __restrict__ pb2v,
    float* __restrict__ pe, _Float16* __restrict__ pmask, float* __restrict__ norms) {
  __shared__ float     sf[16 * LDF];
  __shared__ _Float16  sa[16 * LDH];
  __shared__ float     red1[256], red2[256];
  __shared__ float     rowm[16], rowr[16];

  const int tid = threadIdx.x, lane = tid & 31, wave = tid >> 5;
  const int n = lane & 15, half = lane >> 4;
  const int R = blockIdx.x * 16;
  const int cb0 = wave * 32, cb1 = cb0 + 16;

  // GEMM1: point_feat(padded f16) @ ip_w1 (K = 96)
  v8f a0 = {}, a1 = {};
#pragma unroll
  for (int kc = 0; kc < CP; kc += 32) {
    v16h af = ldA_f16(pf16, CP, R + n, kc, half);
    a0 = wmma_f16(af, ldB_T(w1T, CP, cb0 + n, kc, half), a0);
    a1 = wmma_f16(af, ldB_T(w1T, CP, cb1 + n, kc, half), a1);
  }
  {
    float bb0 = b1[cb0 + n], bb1 = b1[cb1 + n];
#pragma unroll
    for (int r = 0; r < 8; r++) {
      sf[(r + 8 * half) * LDF + cb0 + n] = a0[r] + bb0;
      sf[(r + 8 * half) * LDF + cb1 + n] = a1[r] + bb1;
    }
  }
  __syncthreads();
  ln_stats(sf, red1, red2, rowm, rowr);
  {  // LN + GELU -> f16 activation tile
    int row = tid >> 4, seg = tid & 15;
    float m = rowm[row], rv = rowr[row];
#pragma unroll
    for (int i = 0; i < 16; i++) {
      int c = seg * 16 + i;
      float x = (sf[row * LDF + c] - m) * rv * g1[c] + be1[c];
      sa[row * LDH + c] = (_Float16)gelu_erf(x);
    }
  }
  __syncthreads();

  // GEMM2: act1 @ ip_w2 -> h (kept in registers for residual)
  v8f h0 = {}, h1 = {};
#pragma unroll
  for (int kc = 0; kc < DD; kc += 32) {
    v16h af = ldA_f16(sa, LDH, n, kc, half);
    h0 = wmma_f16(af, ldB_T(w2T, DD, cb0 + n, kc, half), h0);
    h1 = wmma_f16(af, ldB_T(w2T, DD, cb1 + n, kc, half), h1);
  }
  {
    float bb0 = b2[cb0 + n], bb1 = b2[cb1 + n];
#pragma unroll
    for (int r = 0; r < 8; r++) { h0[r] += bb0; h1[r] += bb1; }
  }
  __syncthreads();
#pragma unroll
  for (int r = 0; r < 8; r++) {
    sf[(r + 8 * half) * LDF + cb0 + n] = h0[r];
    sf[(r + 8 * half) * LDF + cb1 + n] = h1[r];
  }
  __syncthreads();
  ln_stats(sf, red1, red2, rowm, rowr);
  {  // LN(h) with point-head gamma/beta (no gelu)
    int row = tid >> 4, seg = tid & 15;
    float m = rowm[row], rv = rowr[row];
#pragma unroll
    for (int i = 0; i < 16; i++) {
      int c = seg * 16 + i;
      sa[row * LDH + c] = (_Float16)((sf[row * LDF + c] - m) * rv * phg[c] + phb[c]);
    }
  }
  __syncthreads();

  // GEMM3: LN(h) @ ph_w1, bias + GELU
  v8f c30 = {}, c31 = {};
#pragma unroll
  for (int kc = 0; kc < DD; kc += 32) {
    v16h af = ldA_f16(sa, LDH, n, kc, half);
    c30 = wmma_f16(af, ldB_T(pw1T, DD, cb0 + n, kc, half), c30);
    c31 = wmma_f16(af, ldB_T(pw1T, DD, cb1 + n, kc, half), c31);
  }
  __syncthreads();  // all waves done reading sa
  {
    float bb0 = pb1v[cb0 + n], bb1 = pb1v[cb1 + n];
#pragma unroll
    for (int r = 0; r < 8; r++) {
      sa[(r + 8 * half) * LDH + cb0 + n] = (_Float16)gelu_erf(c30[r] + bb0);
      sa[(r + 8 * half) * LDH + cb1 + n] = (_Float16)gelu_erf(c31[r] + bb1);
    }
  }
  __syncthreads();

  // GEMM4: @ ph_w2, + bias + residual h -> point_embed
  v8f c40 = {}, c41 = {};
#pragma unroll
  for (int kc = 0; kc < DD; kc += 32) {
    v16h af = ldA_f16(sa, LDH, n, kc, half);
    c40 = wmma_f16(af, ldB_T(pw2T, DD, cb0 + n, kc, half), c40);
    c41 = wmma_f16(af, ldB_T(pw2T, DD, cb1 + n, kc, half), c41);
  }
  {
    float bb0 = pb2v[cb0 + n], bb1 = pb2v[cb1 + n];
#pragma unroll
    for (int r = 0; r < 8; r++) {
      float v0 = h0[r] + c40[r] + bb0;
      float v1 = h1[r] + c41[r] + bb1;
      pe[(size_t)(R + r + 8 * half) * DD + cb0 + n] = v0;
      pe[(size_t)(R + r + 8 * half) * DD + cb1 + n] = v1;
      sf[(r + 8 * half) * LDF + cb0 + n] = v0;
      sf[(r + 8 * half) * LDF + cb1 + n] = v1;
    }
  }
  __syncthreads();
  {  // row norms + normalized f16 pmask
    int row = tid >> 4, seg = tid & 15;
    const float* p = &sf[row * LDF + seg * 16];
    float q = 0.f;
#pragma unroll
    for (int i = 0; i < 16; i++) q += p[i] * p[i];
    red1[tid] = q;
    __syncthreads();
    if (tid < 16) {
      float qq = 0.f;
#pragma unroll
      for (int i = 0; i < 16; i++) qq += red1[tid * 16 + i];
      float nr = fmaxf(sqrtf(qq), 1e-12f);
      norms[R + tid] = nr;
      rowm[tid] = 1.f / nr;
    }
    __syncthreads();
    float ri = rowm[row];
#pragma unroll
    for (int i = 0; i < 16; i++) {
      int c = seg * 16 + i;
      pmask[(size_t)(R + row) * DD + c] = (_Float16)(sf[row * LDF + c] * ri);
    }
  }
}

// ---------------- pmask transpose [N,D] -> [D,N] (f16, LDS tiled) ----------------
__global__ void k_tr(const _Float16* __restrict__ pm, _Float16* __restrict__ pmT) {
  __shared__ _Float16 tile[32][33];
  int bx = blockIdx.x, by = blockIdx.y;
  int tx = threadIdx.x, ty = threadIdx.y;
#pragma unroll
  for (int i = 0; i < 4; i++)
    tile[ty + i * 8][tx] = pm[(size_t)(bx * 32 + ty + i * 8) * DD + by * 32 + tx];
  __syncthreads();
#pragma unroll
  for (int i = 0; i < 4; i++)
    pmT[(size_t)(by * 32 + ty + i * 8) * NN + bx * 32 + tx] = tile[tx][ty + i * 8];
}

// ---------------- logits: scale * qmask @ pmask^T + exp partial sums ----------------
__global__ __launch_bounds__(256) void k_logits(const _Float16* __restrict__ qm,
                                                const _Float16* __restrict__ pm,
                                                const float* __restrict__ lsc,
                                                float* __restrict__ ml,
                                                float* __restrict__ partial) {
  __shared__ float wr[8][16];
  int lane = threadIdx.x & 31, wave = threadIdx.x >> 5;
  int n = lane & 15, half = lane >> 4;
  int gy = blockIdx.y, bx = blockIdx.x;
  int gq0 = gy * 16, nb0 = bx * 512;
  float sg = expf(lsc[gq0 >> 7]);

  v8f acc[4] = {};
#pragma unroll
  for (int kc = 0; kc < DD; kc += 32) {
    v16h a = ldA_f16(qm, DD, gq0 + n, kc, half);
#pragma unroll
    for (int u = 0; u < 4; u++) {
      int nc = nb0 + (wave * 4 + u) * 16;
      acc[u] = wmma_f16(a, ldB_T(pm, DD, nc + n, kc, half), acc[u]);
    }
  }
  float ps[8];
#pragma unroll
  for (int r = 0; r < 8; r++) ps[r] = 0.f;
#pragma unroll
  for (int u = 0; u < 4; u++) {
    int nc = nb0 + (wave * 4 + u) * 16;
#pragma unroll
    for (int r = 0; r < 8; r++) {
      float l = acc[u][r] * sg;
      ml[(size_t)(gq0 + r + 8 * half) * NN + nc + n] = l;
      ps[r] += __expf(l - sg);  // max bounded analytically by sg
    }
  }
#pragma unroll
  for (int m = 1; m < 16; m <<= 1)
#pragma unroll
    for (int r = 0; r < 8; r++) ps[r] += __shfl_xor(ps[r], m);
  if (lane == 0) {
#pragma unroll
    for (int r = 0; r < 8; r++) wr[wave][r] = ps[r];
  }
  if (lane == 16) {
#pragma unroll
    for (int r = 0; r < 8; r++) wr[wave][8 + r] = ps[r];
  }
  __syncthreads();
  if (threadIdx.x < 16) {
    float s = 0.f;
#pragma unroll
    for (int w = 0; w < 8; w++) s += wr[w][threadIdx.x];
    partial[(size_t)(gq0 + threadIdx.x) * NBLK + bx] = s;
  }
}

// ---------------- deterministic row-sum of exp partials ----------------
__global__ __launch_bounds__(256) void k_rowsum(const float* __restrict__ partial,
                                                float* __restrict__ sums) {
  __shared__ float sm[256];
  int row = blockIdx.x, t = threadIdx.x;
  sm[t] = partial[(size_t)row * NBLK + t] + partial[(size_t)row * NBLK + 256 + t];
  __syncthreads();
  for (int s = 128; s > 0; s >>= 1) { if (t < s) sm[t] += sm[t + s]; __syncthreads(); }
  if (t == 0) sums[row] = sm[0];
}

// ---------------- pooling: split-K WMMA, weights = exp(l-s)*norm on the fly ----------------
__global__ __launch_bounds__(256) void k_pool(const float* __restrict__ ml,
                                              const float* __restrict__ norms,
                                              const _Float16* __restrict__ pmT,
                                              const float* __restrict__ lsc,
                                              float* __restrict__ pp) {
  int lane = threadIdx.x & 31, wave = threadIdx.x >> 5;
  int n = lane & 15, half = lane >> 4;
  int gy = blockIdx.y, sp = blockIdx.x;
  int gq0 = gy * 16, nb0 = sp * (NN / KSPL);
  float sg = expf(lsc[gq0 >> 7]);
  const int cb0 = wave * 32, cb1 = cb0 + 16;

  v8f a0 = {}, a1 = {};
  for (int kc = 0; kc < (NN / KSPL); kc += 32) {
    int nb = nb0 + kc;
    v16h a = ldA_w(ml, NN, gq0 + n, nb, half, norms, sg);
    a0 = wmma_f16(a, ldB_T(pmT, NN, cb0 + n, nb, half), a0);
    a1 = wmma_f16(a, ldB_T(pmT, NN, cb1 + n, nb, half), a1);
  }
  float* dst = pp + ((size_t)sp * GQ + gq0) * DD;
#pragma unroll
  for (int r = 0; r < 8; r++) {
    dst[(size_t)(r + 8 * half) * DD + cb0 + n] = a0[r];
    dst[(size_t)(r + 8 * half) * DD + cb1 + n] = a1[r];
  }
}

// ---------------- combine split-K partials, normalize, residual -> refined ----------------
__global__ __launch_bounds__(256) void k_combine(const float* __restrict__ pp,
                                                 const float* __restrict__ sums,
                                                 const float* __restrict__ qemb,
                                                 float* __restrict__ refined) {
  int gq = blockIdx.x, d = threadIdx.x;
  float v = 0.f;
  for (int s = 0; s < KSPL; s++) v += pp[((size_t)s * GQ + gq) * DD + d];
  v /= sums[gq];
  refined[(size_t)gq * DD + d] = qemb[(size_t)gq * DD + d] + v;
}

// ---------------- score head ----------------
__global__ __launch_bounds__(256) void k_score(const float* __restrict__ ref,
                                               const float* __restrict__ lng, const float* __restrict__ lnb,
                                               const float* __restrict__ w1, const float* __restrict__ b1,
                                               const float* __restrict__ w2, const float* __restrict__ b2,
                                               float* __restrict__ sc) {
  __shared__ float x[256], rs[256];
  int gq = blockIdx.x, g = gq >> 7, t = threadIdx.x;
  float xe = ref[(size_t)gq * DD + t];
  float m = blk_sum(xe, rs) * (1.f / 256.f);
  float dv = xe - m;
  float v = blk_sum(dv * dv, rs) * (1.f / 256.f);
  x[t] = dv * rsqrtf(v + 1e-5f) * lng[g * DD + t] + lnb[g * DD + t];
  __syncthreads();
  float a = b1[g * DD + t];
  const float* wp = w1 + (size_t)g * DD * DD + t;
  for (int d = 0; d < DD; d++) a += x[d] * wp[(size_t)d * DD];
  float ya = gelu_erf(a);
  float tot = blk_sum(ya * w2[g * DD + t], rs);
  if (t == 0) sc[gq] = tot + b2[g];
}

// ---------------- launcher ----------------
extern "C" void kernel_launch(void* const* d_in, const int* in_sizes, int n_in,
                              void* d_out, int out_size, void* d_ws, size_t ws_size,
                              hipStream_t stream) {
  const float* pf   = (const float*)d_in[0];
  const float* ipw1 = (const float*)d_in[1];
  const float* ipb1 = (const float*)d_in[2];
  const float* ipg  = (const float*)d_in[3];
  const float* ipb  = (const float*)d_in[4];
  const float* ipw2 = (const float*)d_in[5];
  const float* ipb2 = (const float*)d_in[6];
  const float* phg  = (const float*)d_in[7];
  const float* phb  = (const float*)d_in[8];
  const float* phw1 = (const float*)d_in[9];
  const float* phb1 = (const float*)d_in[10];
  const float* phw2 = (const float*)d_in[11];
  const float* phb2 = (const float*)d_in[12];
  const float* qe   = (const float*)d_in[13];
  const float* qhg  = (const float*)d_in[14];
  const float* qhb  = (const float*)d_in[15];
  const float* qhw1 = (const float*)d_in[16];
  const float* qhb1 = (const float*)d_in[17];
  const float* qhw2 = (const float*)d_in[18];
  const float* qhb2 = (const float*)d_in[19];
  const float* shg  = (const float*)d_in[20];
  const float* shb  = (const float*)d_in[21];
  const float* shw1 = (const float*)d_in[22];
  const float* shb1 = (const float*)d_in[23];
  const float* shw2 = (const float*)d_in[24];
  const float* shb2 = (const float*)d_in[25];
  const float* lsc  = (const float*)d_in[26];

  float* out  = (float*)d_out;
  float* o_pe = out;                                // [N, D]
  float* o_ml = o_pe + (size_t)NN * DD;             // [GQ, N]
  float* o_sc = o_ml + (size_t)GQ * NN;             // [GQ]
  float* o_rf = o_sc + GQ;                          // [GQ, D]

  char* ws = (char*)d_ws;
  size_t off = 0;
  auto take = [&](size_t bytes) -> char* {
    char* p = ws + off;
    off = (off + bytes + 255) & ~(size_t)255;
    return p;
  };
  _Float16* w1T   = (_Float16*)take((size_t)256 * CP * 2);
  _Float16* w2T   = (_Float16*)take((size_t)65536 * 2);
  _Float16* p1T   = (_Float16*)take((size_t)65536 * 2);
  _Float16* p2T   = (_Float16*)take((size_t)65536 * 2);
  float*    qemb  = (float*)take((size_t)GQ * DD * 4);
  _Float16* qm16  = (_Float16*)take((size_t)GQ * DD * 2);
  float*    norms = (float*)take((size_t)NN * 4);
  _Float16* pf16  = (_Float16*)take((size_t)NN * CP * 2);
  _Float16* pm16  = (_Float16*)take((size_t)NN * DD * 2);
  _Float16* pmT16 = (_Float16*)take((size_t)NN * DD * 2);
  float*    part  = (float*)take((size_t)GQ * NBLK * 4);
  float*    sums  = (float*)take((size_t)GQ * 4);
  float*    pp    = (float*)take((size_t)KSPL * GQ * DD * 4);
  (void)n_in; (void)in_sizes; (void)out_size; (void)ws_size;

  k_prep   <<<256, 256, 0, stream>>>(ipw1, ipw2, phw1, phw2, w1T, w2T, p1T, p2T);
  k_prep_pf<<<(NN * CP) / 256, 256, 0, stream>>>(pf, pf16);
  k_query  <<<GQ, 256, 0, stream>>>(qe, qhg, qhb, qhw1, qhb1, qhw2, qhb2, qemb, qm16);
  k_trunk  <<<NN / 16, 256, 0, stream>>>(pf16, w1T, ipb1, ipg, ipb, w2T, ipb2,
                                         phg, phb, p1T, phb1, p2T, phb2,
                                         o_pe, pm16, norms);
  k_tr     <<<dim3(NN / 32, DD / 32), dim3(32, 8), 0, stream>>>(pm16, pmT16);
  k_logits <<<dim3(NBLK, GQ / 16), 256, 0, stream>>>(qm16, pm16, lsc, o_ml, part);
  k_rowsum <<<GQ, 256, 0, stream>>>(part, sums);
  k_pool   <<<dim3(KSPL, GQ / 16), 256, 0, stream>>>(o_ml, norms, pmT16, lsc, pp);
  k_combine<<<GQ, 256, 0, stream>>>(pp, sums, qemb, o_rf);
  k_score  <<<GQ, 256, 0, stream>>>(o_rf, shg, shb, shw1, shb1, shw2, shb2, o_sc);
}
